// I2C_KNN_88862873354498
// MI455X (gfx1250) — compile-verified, matrix operand order checked by reading, not run on hardware
//
#include <hip/hip_runtime.h>
#include <hip/hip_bf16.h>
#include <stdint.h>

typedef __attribute__((ext_vector_type(16))) _Float16     v16h;
typedef __attribute__((ext_vector_type(8)))  float        v8f;
typedef __attribute__((ext_vector_type(4)))  unsigned int v4u;

static constexpr int C        = 640;
static constexpr int HW       = 441;   // 21*21
static constexpr int NB       = 32;    // query images
static constexpr int NS       = 25;    // support images
static constexpr int CLASSES  = 5;
static constexpr int POOL     = 2205;  // shot_num * HW
static constexpr int NTILES   = (POOL + 15) / 16;  // 138 (even)
static constexpr int ROWBLKS  = 7;     // 7 blocks x 64 query rows = 448 >= 441
static constexpr int APAD     = 656;   // padded LDS row stride (halves): 16B-aligned, bank-staggered
static constexpr int SEGS     = C / 8; // 80 x v4u (16B chunks) per 640-half row
static constexpr int SPAD_ROWS = 3;    // Sf over-allocation so tail-tile loads stay in bounds

template <int N> struct IC { static constexpr int v = N; };

// Branchless sorted-insert of v into (t0 >= t1 >= t2). v must not be NaN.
__device__ __forceinline__ void top3_insert(float& t0, float& t1, float& t2, float v) {
    float x = fminf(t0, v);
    t0      = fmaxf(t0, v);
    float y = fminf(t1, x);
    t1      = fmaxf(t1, x);
    t2      = fmaxf(t2, y);
}

// ---------------------------------------------------------------------------
__global__ void zero_out_kernel(float* out, int n) {
    int i = blockIdx.x * blockDim.x + threadIdx.x;
    if (i < n) out[i] = 0.0f;
}

// ---------------------------------------------------------------------------
// L2-normalize each descriptor (over channel dim) and store FP16 row-major
// feature bank: out[b*HW + p][c]. One wave per descriptor.
// ---------------------------------------------------------------------------
__global__ __launch_bounds__(256) void l2norm_f16_kernel(
    const float* __restrict__ x, _Float16* __restrict__ out, int nimg) {
    int gwave = (blockIdx.x * blockDim.x + threadIdx.x) >> 5;
    int lane  = threadIdx.x & 31;
    int ndesc = nimg * HW;
    if (gwave >= ndesc) return;

    int b = gwave / HW;
    int p = gwave % HW;
    const float* base = x + (size_t)b * C * HW + p;

    float v[20];
    float ss = 0.0f;
#pragma unroll
    for (int i = 0; i < 20; ++i) {
        float t = base[(size_t)(lane + i * 32) * HW];
        v[i] = t;
        ss += t * t;
    }
#pragma unroll
    for (int off = 16; off > 0; off >>= 1) ss += __shfl_xor(ss, off, 32);
    float r = rsqrtf(ss);

    _Float16* o = out + (size_t)gwave * C;
#pragma unroll
    for (int i = 0; i < 20; ++i) o[lane + i * 32] = (_Float16)(v[i] * r);
}

// ---------------------------------------------------------------------------
// Main kernel. Block = 128 threads (4 waves). Each wave owns a distinct
// 16-row query tile (64 rows/block in LDS). All waves consume the same
// 16-column B tile, staged in LDS with register double-buffering. Tile loop
// unrolled x2 (compile-time buffer index); last (partial) tile peeled.
// All LDS lives in ONE struct so member order is guaranteed: Bs at offset 0,
// keeping every B-fragment ds_load offset within the unsigned 16-bit
// immediate field (max ~43KB), avoiding per-load VALU address adds.
// ---------------------------------------------------------------------------
struct alignas(16) SharedMem {
    _Float16 Bs[2][16][APAD];     // offset 0      .. 41983
    _Float16 As[64][APAD];        // offset 41984  .. 125951
    float    red[4][32][8][3];    // offset 125952 .. 138239
};

__global__ __launch_bounds__(128) void i2c_knn_kernel(
    const _Float16* __restrict__ Qf, const _Float16* __restrict__ Sf,
    float* __restrict__ out) {
    const int rowBlk = blockIdx.x;  // 0..6
    const int cls    = blockIdx.y;  // 0..4
    const int img    = blockIdx.z;  // 0..31

    __shared__ SharedMem sm;

    const int tid  = threadIdx.x;
    const int wave = tid >> 5;
    const int lane = tid & 31;

    // ---- Stage A (64 rows x 640 halves) into LDS as 16B chunks ----
    for (int i = tid; i < 64 * SEGS; i += 128) {
        int row  = i / SEGS;
        int seg  = i % SEGS;
        int qrow = rowBlk * 64 + row;
        if (qrow > HW - 1) qrow = HW - 1;                    // clamp tail rows
        ((v4u*)&sm.As[row][0])[seg] =
            ((const v4u*)(Qf + ((size_t)img * HW + qrow) * C))[seg];
    }

    // ---- This thread's 10 staging chunks of each 16x640 B tile ----
    // Memory offset of chunk j is 16*j bytes (row stride = 80 chunks), so the
    // global side is affine (one address + immediate offsets).
    int brow[10], bseg[10];
#pragma unroll
    for (int i = 0; i < 10; ++i) {
        int j   = tid + 128 * i;     // 0..1279
        brow[i] = j / SEGS;
        bseg[i] = j % SEGS;
    }

    const _Float16* Sbase = Sf + (size_t)cls * POOL * C;

    // Prefetch B tile 0 into registers.
    v4u breg[10];
#pragma unroll
    for (int i = 0; i < 10; ++i)
        breg[i] = ((const v4u*)(Sbase + (size_t)brow[i] * C))[bseg[i]];

    const int lrow  = lane & 15;        // fragment row (A) / column (B, C/D)
    const int khalf = (lane >> 4) * 8;  // per-half K window base
    const _Float16* arow = &sm.As[wave * 16 + lrow][0];

    float t0[8], t1[8], t2[8];
#pragma unroll
    for (int r = 0; r < 8; ++r) { t0[r] = -1e30f; t1[r] = -1e30f; t2[r] = -1e30f; }

    // One tile step: commit breg -> Bs[P], prefetch tile+1, barrier, 20 WMMAs,
    // branchless top-3 merge. P and MASK are compile-time.
    auto body = [&](auto pc, auto maskc, int tile, bool doPrefetch) {
        constexpr int  P    = decltype(pc)::v;
        constexpr bool MASK = decltype(maskc)::v != 0;
#pragma unroll
        for (int i = 0; i < 10; ++i)
            ((v4u*)&sm.Bs[P][brow[i]][0])[bseg[i]] = breg[i];
        if (doPrefetch) {
#pragma unroll
            for (int i = 0; i < 10; ++i)
                breg[i] = ((const v4u*)(
                    Sbase + (size_t)((tile + 1) * 16 + brow[i]) * C))[bseg[i]];
        }
        __syncthreads();   // Bs[P] visible (A also ready before first use)

        const _Float16* brp = &sm.Bs[P][lrow][0];
        v8f acc = {};
#pragma unroll 4
        for (int k0 = 0; k0 < C; k0 += 32) {
            v16h a, b;
            *((v4u*)&a)     = *((const v4u*)(arow + k0 + khalf));
            *((v4u*)&a + 1) = *((const v4u*)(arow + k0 + khalf + 16));
            *((v4u*)&b)     = *((const v4u*)(brp + k0 + khalf));
            *((v4u*)&b + 1) = *((const v4u*)(brp + k0 + khalf + 16));
            acc = __builtin_amdgcn_wmma_f32_16x16x32_f16(
                false, a, false, b, (short)0, acc, false, false);
        }

        if constexpr (MASK) {
            const bool ok = (tile * 16 + lrow) < POOL;   // full cndmask select:
#pragma unroll                                           // NaN/Inf garbage cannot leak
            for (int r = 0; r < 8; ++r)
                top3_insert(t0[r], t1[r], t2[r], ok ? acc[r] : -1e30f);
        } else {
#pragma unroll
            for (int r = 0; r < 8; ++r)
                top3_insert(t0[r], t1[r], t2[r], acc[r]);
        }
        // Next step writes the other buffer; re-write of this buffer happens
        // only after the next barrier, so one barrier per tile is race-free.
    };

    // Tiles 0..135: fully in-range, unrolled x2 for constant buffer index.
    for (int tile = 0; tile < NTILES - 2; tile += 2) {
        body(IC<0>{}, IC<0>{}, tile, true);
        body(IC<1>{}, IC<0>{}, tile + 1, true);
    }
    body(IC<0>{}, IC<0>{}, NTILES - 2, true);    // tile 136 (in-range)
    body(IC<1>{}, IC<1>{}, NTILES - 1, false);   // tile 137 (masked tail)

#pragma unroll
    for (int r = 0; r < 8; ++r) {
        sm.red[wave][lane][r][0] = t0[r];
        sm.red[wave][lane][r][1] = t1[r];
        sm.red[wave][lane][r][2] = t2[r];
    }
    __syncthreads();

    // ---- Wave-local final merge: rows belong to this wave only ----
    {
        const int m     = lane & 15;           // row within wave's 16-row tile
        const int r     = m & 7;
        const int lbase = (m >> 3) * 16;       // lanes holding this row
        float b0 = -1e30f, b1 = -1e30f, b2 = -1e30f;
        for (int l = lbase; l < lbase + 16; ++l) {
            top3_insert(b0, b1, b2, sm.red[wave][l][r][0]);
            top3_insert(b0, b1, b2, sm.red[wave][l][r][1]);
            top3_insert(b0, b1, b2, sm.red[wave][l][r][2]);
        }
        const int qrow = rowBlk * 64 + wave * 16 + m;
        float sum = (lane < 16 && qrow < HW) ? (b0 + b1 + b2) : 0.0f;
#pragma unroll
        for (int off = 16; off > 0; off >>= 1) sum += __shfl_xor(sum, off, 32);
        if (lane == 0) atomicAdd(&out[img * CLASSES + cls], sum);
    }
}

// ---------------------------------------------------------------------------
extern "C" void kernel_launch(void* const* d_in, const int* in_sizes, int n_in,
                              void* d_out, int out_size, void* d_ws, size_t ws_size,
                              hipStream_t stream) {
    (void)in_sizes; (void)n_in; (void)ws_size;
    const float* q = (const float*)d_in[0];   // [32,640,21,21]
    const float* S = (const float*)d_in[1];   // [25,640,21,21]
    // d_in[2] = shot_num (device scalar) -> fixed to 5 per setup_inputs

    _Float16* Qf = (_Float16*)d_ws;                       // 14112 x 640 f16
    _Float16* Sf = Qf + (size_t)NB * HW * C;              // (11025+SPAD_ROWS) x 640 f16
    // Kernel may read up to SPAD_ROWS garbage rows past the 11025 written
    // rows of Sf; those columns are mask-selected out with a full cndmask,
    // so NaN/Inf garbage cannot reach the top-3 state.

    float* out = (float*)d_out;
    zero_out_kernel<<<(out_size + 255) / 256, 256, 0, stream>>>(out, out_size);

    {
        int ndesc = NB * HW;
        l2norm_f16_kernel<<<(ndesc + 7) / 8, 256, 0, stream>>>(q, Qf, NB);
    }
    {
        int ndesc = NS * HW;
        l2norm_f16_kernel<<<(ndesc + 7) / 8, 256, 0, stream>>>(S, Sf, NS);
    }

    dim3 grid(ROWBLKS, CLASSES, NB);    // 7 x 5 x 32 = 1120 blocks
    i2c_knn_kernel<<<grid, 128, 0, stream>>>(Qf, Sf, out);
}
